// Encoder_83794811945678
// MI455X (gfx1250) — matrix operand order repaired and checked by reading, not compiled, not run on hardware
//
#include <hip/hip_runtime.h>
#include <stdint.h>

// ---------------- problem constants (from reference) ----------------
#define B_    2
#define NG    32768
#define NM    10242
#define CIN   64
#define HD    128
#define EG2M  163840
#define EM2M  61440
#define EM2G  98304
#define KP_GRID  96     // 67 padded to multiple of 32
#define KP_SMALL 32     // 3/4 padded to 32

// ---------------- WMMA types ----------------
typedef __attribute__((ext_vector_type(16))) __bf16 v16bf;
typedef __attribute__((ext_vector_type(8)))  float  v8f;

union BfFrag { v16bf v; unsigned u32[8]; };

__device__ __forceinline__ unsigned short f2bf(float f) {
  unsigned u = __float_as_uint(f);
  unsigned r = ((u >> 16) & 1u) + 0x7FFFu;   // round-to-nearest-even
  return (unsigned short)((u + r) >> 16);
}
__device__ __forceinline__ unsigned pack2bf(float a, float b) {
  return (unsigned)f2bf(a) | ((unsigned)f2bf(b) << 16);
}

// LDS layouts (dword units), chosen for 16B alignment + conflict-free b128 reads
#define APW 68   // A: [32 m][64 kpair + pad]  (lane stride 68 % 64 = 4 -> distinct banks)
#define BPW 20   // B: [128 n][16 kpair + pad] (lane stride 20 % 64 -> distinct banks)
#define HP  132  // hidden f32 pitch

// ============================================================================
// Fused MLP: out[rows,128] = (res?) + LN(SiLU(X@W1+b1))*g+beta @ W2 + b2
// X: packed bf16 [rows, Kin], Kin % 32 == 0, valid columns = d1 (rest zero).
// 256 threads = 8 waves; block covers 32 rows x 128 cols.
// wave w: M-tile = w&1, N-tiles = 2*(w>>1), 2*(w>>1)+1.
// ============================================================================
__launch_bounds__(256)
__global__ void k_mlp_fused(const unsigned short* __restrict__ X, int Kin, int d1,
                            const float* __restrict__ W1, const float* __restrict__ b1,
                            const float* __restrict__ g,  const float* __restrict__ beta,
                            const float* __restrict__ W2, const float* __restrict__ b2,
                            const float* __restrict__ res,
                            float* __restrict__ out, long rows)
{
  __shared__ unsigned lds_am[32 * APW];   // A operand, packed bf16 pairs [m][kpair]
  __shared__ unsigned lds_bn[128 * BPW];  // B operand, packed bf16 pairs [n][kpair]
  __shared__ float    lds_h[32 * HP];     // hidden activations (f32)
  __shared__ float    lds_ps[32 * 8], lds_ps2[32 * 8];
  __shared__ float    lds_mu[32], lds_rs[32];

  const int  t    = threadIdx.x;
  const int  lane = t & 31;
  const int  wave = t >> 5;
  const int  mt   = wave & 1;        // M tile (0/1)
  const int  ng   = wave >> 1;       // N group (0..3) -> tiles 2ng, 2ng+1
  const long row0 = (long)blockIdx.x * 32;
  const int  m16  = lane & 15;
  const int  hi   = lane >> 4;
  const int  n0   = (ng * 2) * 16 + m16;
  const int  n1   = n0 + 16;

  v8f acc0, acc1;
  for (int r = 0; r < 8; ++r) { acc0[r] = 0.f; acc1[r] = 0.f; }

  // ---------------- GEMM1: h = X @ W1 ----------------
  for (int kk = 0; kk < Kin; kk += 32) {
    __syncthreads();

    // stage X tile [32 rows][32 k]: global bf16 pairs map byte-for-byte onto
    // packed A layout -> async DMA 16B per lane (threads 0..127 cover the tile)
    if (t < 128) {
      int  r  = t >> 2;             // 0..31
      int  c8 = (t & 3) * 8;        // 0,8,16,24 (8 bf16 = 16B)
      long rr = row0 + r;
      unsigned* dst = &lds_am[r * APW + (c8 >> 1)];
      if (rr < rows) {
        unsigned lds_off = (unsigned)(uintptr_t)dst;
        unsigned long long gaddr =
            (unsigned long long)(uintptr_t)(X + rr * (long)Kin + kk + c8);
        asm volatile("global_load_async_to_lds_b128 %0, %1, off"
                     :: "v"(lds_off), "v"(gaddr) : "memory");
      } else {
        uint4 z; z.x = 0; z.y = 0; z.z = 0; z.w = 0;
        *(uint4*)dst = z;
      }
    }

    // stage W1 tile [32 k][128 n] as packed pairs [n][kpair]
    for (int it = 0; it < 2; ++it) {
      int idx = t + it * 256;         // 0..511
      int kp  = idx & 15;             // kpair 0..15
      int nn  = (idx >> 4) * 4;       // n 0..124 step 4
      int k0  = kk + kp * 2;
      float4 w0, w1;
      if (k0 < d1)     w0 = *(const float4*)&W1[(long)k0 * HD + nn];
      else             { w0.x = w0.y = w0.z = w0.w = 0.f; }
      if (k0 + 1 < d1) w1 = *(const float4*)&W1[(long)(k0 + 1) * HD + nn];
      else             { w1.x = w1.y = w1.z = w1.w = 0.f; }
      lds_bn[(nn + 0) * BPW + kp] = pack2bf(w0.x, w1.x);
      lds_bn[(nn + 1) * BPW + kp] = pack2bf(w0.y, w1.y);
      lds_bn[(nn + 2) * BPW + kp] = pack2bf(w0.z, w1.z);
      lds_bn[(nn + 3) * BPW + kp] = pack2bf(w0.w, w1.w);
      // prefetch next K tile of W1 into caches while this tile computes
      if (it == 0 && k0 + 32 < d1)
        __builtin_prefetch(&W1[(long)(k0 + 32) * HD + nn], 0, 1);
    }

    asm volatile("s_wait_asynccnt 0x0" ::: "memory");
    __syncthreads();

    // A fragment: two b128 LDS reads  (u32[p] = kpair (p>>2)*8 + hi*4 + (p&3))
    const unsigned* arow = &lds_am[(mt * 16 + m16) * APW];
    BfFrag af;
    {
      uint4 lo = *(const uint4*)(arow + hi * 4);
      uint4 hh = *(const uint4*)(arow + 8 + hi * 4);
      af.u32[0] = lo.x; af.u32[1] = lo.y; af.u32[2] = lo.z; af.u32[3] = lo.w;
      af.u32[4] = hh.x; af.u32[5] = hh.y; af.u32[6] = hh.z; af.u32[7] = hh.w;
    }
    // B fragments: two b128 LDS reads each (u32[p] = kpair (p>>2)*8 + hi*4 + (p&3))
    BfFrag bf0, bf1;
    {
      const unsigned* b0 = &lds_bn[n0 * BPW + hi * 4];
      uint4 lo = *(const uint4*)b0;
      uint4 hh = *(const uint4*)(b0 + 8);
      bf0.u32[0] = lo.x; bf0.u32[1] = lo.y; bf0.u32[2] = lo.z; bf0.u32[3] = lo.w;
      bf0.u32[4] = hh.x; bf0.u32[5] = hh.y; bf0.u32[6] = hh.z; bf0.u32[7] = hh.w;
      const unsigned* b1p = &lds_bn[n1 * BPW + hi * 4];
      uint4 lo1 = *(const uint4*)b1p;
      uint4 hh1 = *(const uint4*)(b1p + 8);
      bf1.u32[0] = lo1.x; bf1.u32[1] = lo1.y; bf1.u32[2] = lo1.z; bf1.u32[3] = lo1.w;
      bf1.u32[4] = hh1.x; bf1.u32[5] = hh1.y; bf1.u32[6] = hh1.z; bf1.u32[7] = hh1.w;
    }
    acc0 = __builtin_amdgcn_wmma_f32_16x16x32_bf16(false, af.v, false, bf0.v,
                                                   (short)0, acc0, false, false);
    acc1 = __builtin_amdgcn_wmma_f32_16x16x32_bf16(false, af.v, false, bf1.v,
                                                   (short)0, acc1, false, false);
  }
  __syncthreads();

  // bias + SiLU -> lds_h   (C layout: VGPR r -> M = hi*8 + r, N = lane&15)
  {
    const float bb0 = b1[n0], bb1 = b1[n1];
    #pragma unroll
    for (int r = 0; r < 8; ++r) {
      int mr = mt * 16 + hi * 8 + r;
      float h0 = acc0[r] + bb0;
      float h1 = acc1[r] + bb1;
      lds_h[mr * HP + n0] = h0 / (1.f + __expf(-h0));
      lds_h[mr * HP + n1] = h1 / (1.f + __expf(-h1));
    }
  }
  __syncthreads();

  // per-row LayerNorm stats, two-phase (8 partials per row)
  {
    int r = t >> 3, seg = t & 7;
    float s = 0.f, s2 = 0.f;
    const float* hrow = &lds_h[r * HP + seg * 16];
    #pragma unroll
    for (int c = 0; c < 16; ++c) { float v = hrow[c]; s += v; s2 += v * v; }
    lds_ps[r * 8 + seg] = s; lds_ps2[r * 8 + seg] = s2;
  }
  __syncthreads();
  if (t < 32) {
    float s = 0.f, s2 = 0.f;
    #pragma unroll
    for (int j = 0; j < 8; ++j) { s += lds_ps[t * 8 + j]; s2 += lds_ps2[t * 8 + j]; }
    float mu  = s * (1.f / HD);
    float var = s2 * (1.f / HD) - mu * mu;
    lds_mu[t] = mu;
    lds_rs[t] = rsqrtf(var + 1e-5f);
  }
  __syncthreads();

  // normalize + affine -> packed bf16 pairs in lds_am (absolute kpairs 0..63)
  for (int i = t; i < 32 * 32; i += 256) {
    int r = i >> 5, c4 = (i & 31) * 4;
    float mu = lds_mu[r], rs = lds_rs[r];
    const float* hrow = &lds_h[r * HP + c4];
    float v0 = (hrow[0] - mu) * rs * g[c4 + 0] + beta[c4 + 0];
    float v1 = (hrow[1] - mu) * rs * g[c4 + 1] + beta[c4 + 1];
    float v2 = (hrow[2] - mu) * rs * g[c4 + 2] + beta[c4 + 2];
    float v3 = (hrow[3] - mu) * rs * g[c4 + 3] + beta[c4 + 3];
    unsigned long long pk = (unsigned long long)pack2bf(v0, v1)
                          | ((unsigned long long)pack2bf(v2, v3) << 32);
    *(unsigned long long*)&lds_am[r * APW + (c4 >> 1)] = pk;
  }

  for (int r = 0; r < 8; ++r) { acc0[r] = 0.f; acc1[r] = 0.f; }

  // ---------------- GEMM2: out = A2 @ W2 ----------------
  for (int kk = 0; kk < HD; kk += 32) {
    __syncthreads();
    for (int it = 0; it < 2; ++it) {
      int idx = t + it * 256;
      int kp  = idx & 15;
      int nn  = (idx >> 4) * 4;
      int k0  = kk + kp * 2;
      float4 w0 = *(const float4*)&W2[(long)k0 * HD + nn];
      float4 w1 = *(const float4*)&W2[(long)(k0 + 1) * HD + nn];
      lds_bn[(nn + 0) * BPW + kp] = pack2bf(w0.x, w1.x);
      lds_bn[(nn + 1) * BPW + kp] = pack2bf(w0.y, w1.y);
      lds_bn[(nn + 2) * BPW + kp] = pack2bf(w0.z, w1.z);
      lds_bn[(nn + 3) * BPW + kp] = pack2bf(w0.w, w1.w);
      if (it == 0 && k0 + 32 < HD)
        __builtin_prefetch(&W2[(long)(k0 + 32) * HD + nn], 0, 1);
    }
    __syncthreads();

    const unsigned* arow = &lds_am[(mt * 16 + m16) * APW + (kk >> 1)];
    BfFrag af;
    {
      uint4 lo = *(const uint4*)(arow + hi * 4);
      uint4 hh = *(const uint4*)(arow + 8 + hi * 4);
      af.u32[0] = lo.x; af.u32[1] = lo.y; af.u32[2] = lo.z; af.u32[3] = lo.w;
      af.u32[4] = hh.x; af.u32[5] = hh.y; af.u32[6] = hh.z; af.u32[7] = hh.w;
    }
    BfFrag bf0, bf1;
    {
      const unsigned* b0 = &lds_bn[n0 * BPW + hi * 4];
      uint4 lo = *(const uint4*)b0;
      uint4 hh = *(const uint4*)(b0 + 8);
      bf0.u32[0] = lo.x; bf0.u32[1] = lo.y; bf0.u32[2] = lo.z; bf0.u32[3] = lo.w;
      bf0.u32[4] = hh.x; bf0.u32[5] = hh.y; bf0.u32[6] = hh.z; bf0.u32[7] = hh.w;
      const unsigned* b1p = &lds_bn[n1 * BPW + hi * 4];
      uint4 lo1 = *(const uint4*)b1p;
      uint4 hh1 = *(const uint4*)(b1p + 8);
      bf1.u32[0] = lo1.x; bf1.u32[1] = lo1.y; bf1.u32[2] = lo1.z; bf1.u32[3] = lo1.w;
      bf1.u32[4] = hh1.x; bf1.u32[5] = hh1.y; bf1.u32[6] = hh1.z; bf1.u32[7] = hh1.w;
    }
    acc0 = __builtin_amdgcn_wmma_f32_16x16x32_bf16(false, af.v, false, bf0.v,
                                                   (short)0, acc0, false, false);
    acc1 = __builtin_amdgcn_wmma_f32_16x16x32_bf16(false, af.v, false, bf1.v,
                                                   (short)0, acc1, false, false);
  }

  // epilogue: + b2 (+ residual), store f32
  {
    const float bb0 = b2[n0], bb1 = b2[n1];
    #pragma unroll
    for (int r = 0; r < 8; ++r) {
      long mr = row0 + mt * 16 + hi * 8 + r;
      if (mr < rows) {
        float v0 = acc0[r] + bb0;
        float v1 = acc1[r] + bb1;
        if (res) { v0 += res[mr * HD + n0]; v1 += res[mr * HD + n1]; }
        out[mr * HD + n0] = v0;
        out[mr * HD + n1] = v1;
      }
    }
  }
}

// ============================================================================
// Input builders / glue kernels
// ============================================================================
__global__ void k_build_grid_in(const float* __restrict__ inp, const float* __restrict__ gs,
                                unsigned short* __restrict__ out, long total)
{
  long i = (long)blockIdx.x * 256 + threadIdx.x;
  if (i >= total) return;
  int  c   = (int)(i % KP_GRID);
  long row = i / KP_GRID;
  int  n   = (int)(row % NG);
  long b   = row / NG;
  float v = 0.f;
  if (c < CIN)          v = inp[(b * NG + n) * CIN + c];
  else if (c < CIN + 3) v = gs[(long)n * 3 + (c - CIN)];
  out[i] = f2bf(v);
}

__global__ void k_bcast_pad(const float* __restrict__ src, unsigned short* __restrict__ out,
                            long nsrc, int csrc, long total)
{
  long i = (long)blockIdx.x * 256 + threadIdx.x;
  if (i >= total) return;
  int  c   = (int)(i % KP_SMALL);
  long row = i / KP_SMALL;
  long sr  = row % nsrc;
  float v = (c < csrc) ? src[sr * csrc + c] : 0.f;
  out[i] = f2bf(v);
}

__global__ void k_build_edge_in(const float* __restrict__ eg, const float* __restrict__ vG,
                                const float* __restrict__ vM, const int* __restrict__ snd,
                                const int* __restrict__ rcv, unsigned short* __restrict__ out,
                                long total)
{
  long i = (long)blockIdx.x * 256 + threadIdx.x;
  if (i >= total) return;
  int  c   = (int)(i % (3 * HD));
  long row = i / (3 * HD);
  long e   = row % EG2M;
  long b   = row / EG2M;
  float v;
  if (c < HD)            v = eg[row * HD + c];
  else if (c < 2 * HD)   v = vG[(b * NG + snd[e]) * (long)HD + (c - HD)];
  else                   v = vM[(b * NM + rcv[e]) * (long)HD + (c - 2 * HD)];
  out[i] = f2bf(v);
}

__global__ void k_build_nu_in(const float* __restrict__ vM, const float* __restrict__ agg,
                              unsigned short* __restrict__ out, long total)
{
  long i = (long)blockIdx.x * 256 + threadIdx.x;
  if (i >= total) return;
  int  c   = (int)(i % (2 * HD));
  long row = i / (2 * HD);
  float v = (c < HD) ? vM[row * HD + c] : agg[row * HD + (c - HD)];
  out[i] = f2bf(v);
}

__global__ void k_f32_bf16(const float* __restrict__ src, unsigned short* __restrict__ dst, long n)
{
  long i = (long)blockIdx.x * 256 + threadIdx.x;
  if (i < n) dst[i] = f2bf(src[i]);
}

__global__ void k_zero(float* __restrict__ p, long n)
{
  long i = (long)blockIdx.x * 256 + threadIdx.x;
  if (i < n) p[i] = 0.f;
}

__global__ void k_scatter_add(const float* __restrict__ e, const int* __restrict__ rcv,
                              float* __restrict__ agg, long total)
{
  long i = (long)blockIdx.x * 256 + threadIdx.x;
  if (i >= total) return;
  int  h   = (int)(i & (HD - 1));
  long row = i >> 7;
  long ee  = row % EG2M;
  long b   = row / EG2M;
  atomicAdd(&agg[(b * NM + rcv[ee]) * (long)HD + h], e[i]);
}

// ============================================================================
// Orchestration
// ============================================================================
extern "C" void kernel_launch(void* const* d_in, const int* in_sizes, int n_in,
                              void* d_out, int out_size, void* d_ws, size_t ws_size,
                              hipStream_t stream)
{
  (void)in_sizes; (void)n_in; (void)out_size; (void)ws_size;

  const float* inp  = (const float*)d_in[0];
  const float* gstr = (const float*)d_in[1];
  const float* mstr = (const float*)d_in[2];
  const float* m2ms = (const float*)d_in[3];
  const float* g2ms = (const float*)d_in[4];
  const float* m2gs = (const float*)d_in[5];
  const int*   snd  = (const int*)d_in[6];
  const int*   rcv  = (const int*)d_in[7];
  // params in dict insertion order: 8 MLPs x (W1,b1,g,beta,W2,b2)
  const float* const* P = (const float* const*)(d_in + 8);
  const int GRID_E = 0, MESH_E = 6, M2M_E = 12, G2M_E = 18, M2G_E = 24,
            MSG = 30, NODE = 36, GUPD = 42;

  float* out  = (float*)d_out;
  float* vM   = out;                                   // [B*NM, H]
  float* vG   = out  + (long)B_ * NM * HD;             // [B*NG, H]
  float* eM2M = vG   + (long)B_ * NG * HD;             // [EM2M, H]
  float* eM2G = eM2M + (long)EM2M * HD;                // [EM2G, H]

  // workspace carve
  char* ws = (char*)d_ws;
  size_t off = 0;
  auto take = [&](size_t bytes) -> void* {
    void* p = ws + off;
    off += (bytes + 255) & ~(size_t)255;
    return p;
  };
  float*          ws_e   = (float*)take((size_t)B_ * EG2M * HD * 4);          // eG2M
  unsigned short* ws_xg  = (unsigned short*)take((size_t)B_ * NG * KP_GRID * 2);
  unsigned short* ws_xm  = (unsigned short*)take((size_t)B_ * NM * KP_SMALL * 2);
  unsigned short* ws_xe  = (unsigned short*)take((size_t)B_ * EG2M * KP_SMALL * 2);
  size_t edge_off = off;
  unsigned short* ws_ein = (unsigned short*)take((size_t)B_ * EG2M * 3 * HD * 2);
  (void)ws_ein;
  // aliases into edge_in region, used only after edge_in is dead
  float*          ws_agg = (float*)(ws + edge_off);
  unsigned short* ws_nu  = (unsigned short*)(ws + edge_off +
                              (((size_t)B_ * NM * HD * 4 + 255) & ~(size_t)255));
  unsigned short* ws_vgb = (unsigned short*)((char*)ws_nu +
                              (((size_t)B_ * NM * 2 * HD * 2 + 255) & ~(size_t)255));

  auto mlp = [&](const unsigned short* X, int Kin, int d1, int pb,
                 const float* res, float* o, long rows) {
    k_mlp_fused<<<dim3((unsigned)((rows + 31) / 32)), dim3(256), 0, stream>>>(
        X, Kin, d1, P[pb + 0], P[pb + 1], P[pb + 2], P[pb + 3], P[pb + 4], P[pb + 5],
        res, o, rows);
  };
  auto blocks = [](long n) { return dim3((unsigned)((n + 255) / 256)); };

  // 1) grid node embed -> vG
  {
    long tot = (long)B_ * NG * KP_GRID;
    k_build_grid_in<<<blocks(tot), 256, 0, stream>>>(inp, gstr, ws_xg, tot);
    mlp(ws_xg, KP_GRID, CIN + 3, GRID_E, nullptr, vG, (long)B_ * NG);
  }
  // 2) mesh node embed -> vM
  {
    long tot = (long)B_ * NM * KP_SMALL;
    k_bcast_pad<<<blocks(tot), 256, 0, stream>>>(mstr, ws_xm, NM, 3, tot);
    mlp(ws_xm, KP_SMALL, 3, MESH_E, nullptr, vM, (long)B_ * NM);
  }
  // 3) G2M edge embed -> ws_e
  {
    long tot = (long)B_ * EG2M * KP_SMALL;
    k_bcast_pad<<<blocks(tot), 256, 0, stream>>>(g2ms, ws_xe, EG2M, 4, tot);
    mlp(ws_xe, KP_SMALL, 4, G2M_E, nullptr, ws_e, (long)B_ * EG2M);
  }
  // 4) edge message: gather/concat -> MLP with residual into ws_e
  {
    long tot = (long)B_ * EG2M * 3 * HD;
    k_build_edge_in<<<blocks(tot), 256, 0, stream>>>(ws_e, vG, vM, snd, rcv, ws_ein, tot);
    mlp(ws_ein, 3 * HD, 3 * HD, MSG, ws_e, ws_e, (long)B_ * EG2M);
  }
  // 5) scatter-add into agg
  {
    long na = (long)B_ * NM * HD;
    k_zero<<<blocks(na), 256, 0, stream>>>(ws_agg, na);
    long tot = (long)B_ * EG2M * HD;
    k_scatter_add<<<blocks(tot), 256, 0, stream>>>(ws_e, rcv, ws_agg, tot);
  }
  // 6) mesh node update: vM += MLP([vM, agg])
  {
    long tot = (long)B_ * NM * 2 * HD;
    k_build_nu_in<<<blocks(tot), 256, 0, stream>>>(vM, ws_agg, ws_nu, tot);
    mlp(ws_nu, 2 * HD, 2 * HD, NODE, vM, vM, (long)B_ * NM);
  }
  // 7) grid update: vG += MLP(vG)
  {
    long n = (long)B_ * NG * HD;
    k_f32_bf16<<<blocks(n), 256, 0, stream>>>(vG, ws_vgb, n);
    mlp(ws_vgb, HD, HD, GUPD, vG, vG, (long)B_ * NG);
  }
  // 8) M2M edge embed -> eM2M
  {
    long tot = (long)EM2M * KP_SMALL;
    k_bcast_pad<<<blocks(tot), 256, 0, stream>>>(m2ms, ws_xe, EM2M, 4, tot);
    mlp(ws_xe, KP_SMALL, 4, M2M_E, nullptr, eM2M, (long)EM2M);
  }
  // 9) M2G edge embed -> eM2G
  {
    long tot = (long)EM2G * KP_SMALL;
    k_bcast_pad<<<blocks(tot), 256, 0, stream>>>(m2gs, ws_xe, EM2G, 4, tot);
    mlp(ws_xe, KP_SMALL, 4, M2G_E, nullptr, eM2G, (long)EM2G);
  }
}
